// MultiBoxLoss_34007551050158
// MI455X (gfx1250) — compile-verified, matrix operand order checked by reading, not run on hardware
//
#include <hip/hip_runtime.h>
#include <hip/hip_bf16.h>
#include <math.h>

// Problem constants (match reference)
#define BB      16
#define PP      19248
#define NOBJ    32
#define NC      81
#define POS_TH  0.5f
#define NEG_TH  0.4f
#define VAR0    0.1f
#define VAR1    0.2f

typedef __attribute__((ext_vector_type(16))) _Float16 v16h;
typedef __attribute__((ext_vector_type(8)))  float    v8f;

union V16H { v16h v; _Float16 e[16]; };
union V8F  { v8f  v; float    e[8];  };

// ---------------- K0: init accumulators / scratch ----------------
__global__ void k0_init(float* lossAcc, int* nPos, int* Ntot,
                        unsigned long long* bestPk, unsigned int* hist) {
    int i = blockIdx.x * 256 + threadIdx.x;
    if (i < 2)          lossAcc[i] = 0.0f;
    if (i < BB)         nPos[i] = 0;
    if (i == 0)         *Ntot = 0;
    if (i < BB * NOBJ)  bestPk[i] = 0xFFFFFFFFull;      // iou=0, prior=0 (first-index tie rule)
    if (i < BB * 1024)  hist[i] = 0u;
}

// ---------------- K1: IoU matching ----------------
__global__ void k1_match(const float* __restrict__ priors,
                         const float* __restrict__ gtb,
                         float* __restrict__ btOv, int* __restrict__ btIdx,
                         unsigned long long* __restrict__ bestPk) {
    __shared__ float sG[NOBJ][4];
    __shared__ unsigned long long sB[NOBJ];
    int b = blockIdx.y, t = threadIdx.x;
    if (t < NOBJ * 4) ((float*)sG)[t] = gtb[b * NOBJ * 4 + t];
    if (t < NOBJ)     sB[t] = 0ull;
    __syncthreads();

    int p = blockIdx.x * 256 + t;
    bool act = p < PP;
    int pc = act ? p : PP - 1;
    float cx = priors[pc*4+0], cy = priors[pc*4+1];
    float pw = priors[pc*4+2], ph = priors[pc*4+3];
    float px0 = cx - pw*0.5f, py0 = cy - ph*0.5f;
    float px1 = cx + pw*0.5f, py1 = cy + ph*0.5f;
    float areaP = pw * ph;
    float bestOv = -1.0f; int bestN = 0;
    unsigned pkey = 0xFFFFFFFFu - (unsigned)p;   // larger key == smaller prior index

    for (int n = 0; n < NOBJ; ++n) {
        float g0 = sG[n][0], g1 = sG[n][1], g2 = sG[n][2], g3 = sG[n][3];
        float ix = fmaxf(fminf(px1, g2) - fmaxf(px0, g0), 0.0f);
        float iy = fmaxf(fminf(py1, g3) - fmaxf(py0, g1), 0.0f);
        float inter = ix * iy;
        float areaG = (g2 - g0) * (g3 - g1);
        float iou = inter / (areaG + areaP - inter);
        if (iou > bestOv) { bestOv = iou; bestN = n; }
        if (act) {
            unsigned long long key =
                ((unsigned long long)__float_as_uint(iou) << 32) | pkey;
            atomicMax(&sB[n], key);
        }
    }
    if (act) {
        size_t i = (size_t)b * PP + p;
        btOv[i] = bestOv; btIdx[i] = bestN;
    }
    __syncthreads();
    if (t < NOBJ) atomicMax(&bestPk[b * NOBJ + t], sB[t]);
}

// ---------------- K2: override + conf_t + smooth-L1 loc loss ----------------
__global__ void k2_loc(const float* __restrict__ loc,
                       const float* __restrict__ priors,
                       const float* __restrict__ gtb,
                       const int*   __restrict__ gtl,
                       const float* __restrict__ btOv,
                       const int*   __restrict__ btIdx,
                       const unsigned long long* __restrict__ bestPk,
                       int* __restrict__ confT,
                       float* lossAcc, int* nPos, int* Ntot) {
    __shared__ float sG[NOBJ][4];
    __shared__ int sL[NOBJ];
    __shared__ unsigned sBP[NOBJ];
    __shared__ float sRed[256];
    __shared__ int sCnt[256];
    int b = blockIdx.y, t = threadIdx.x;
    if (t < NOBJ * 4) ((float*)sG)[t] = gtb[b * NOBJ * 4 + t];
    if (t < NOBJ) {
        sL[t]  = gtl[b * NOBJ + t];
        sBP[t] = 0xFFFFFFFFu - (unsigned)(bestPk[b * NOBJ + t] & 0xFFFFFFFFull);
    }
    __syncthreads();

    int p = blockIdx.x * 256 + t;
    float l1 = 0.0f; int cnt = 0;
    if (p < PP) {
        size_t i = (size_t)b * PP + p;
        float ov = btOv[i]; int idx = btIdx[i];
        for (int n = 0; n < NOBJ; ++n)
            if (sBP[n] == (unsigned)p) { idx = n; ov = 2.0f; }
        int c = sL[idx] + 1;
        if (ov < POS_TH) c = -1;
        if (ov < NEG_TH) c = 0;
        confT[i] = c;
        if (c > 0) {
            cnt = 1;
            float g0 = sG[idx][0], g1 = sG[idx][1], g2 = sG[idx][2], g3 = sG[idx][3];
            float cx = priors[p*4+0], cy = priors[p*4+1];
            float pw = priors[p*4+2], ph = priors[p*4+3];
            float mcx = (g0+g2)*0.5f, mcy = (g1+g3)*0.5f;
            float mw = g2 - g0, mh = g3 - g1;
            float enc[4];
            enc[0] = (mcx - cx) / (VAR0 * pw);
            enc[1] = (mcy - cy) / (VAR0 * ph);
            enc[2] = logf(mw / pw) / VAR1;
            enc[3] = logf(mh / ph) / VAR1;
            for (int k = 0; k < 4; ++k) {
                float d = loc[i*4 + k] - enc[k];
                float ad = fabsf(d);
                l1 += (ad < 1.0f) ? 0.5f * d * d : ad - 0.5f;
            }
        }
    }
    sRed[t] = l1; sCnt[t] = cnt;
    __syncthreads();
    for (int s = 128; s > 0; s >>= 1) {
        if (t < s) { sRed[t] += sRed[t + s]; sCnt[t] += sCnt[t + s]; }
        __syncthreads();
    }
    if (t == 0) {
        if (sRed[0] != 0.0f) atomicAdd(lossAcc + 0, sRed[0]);
        if (sCnt[0]) { atomicAdd(&nPos[b], sCnt[0]); atomicAdd(Ntot, sCnt[0]); }
    }
}

// ---------------- K3: async-LDS staging + WMMA log-softmax / NLL ----------------
// One wave32 per 16 priors (8 waves/block). Each wave stages its 16 conf rows
// into LDS with GLOBAL_LOAD_ASYNC_TO_LDS_B128 (ASYNCcnt path): each 324B row is
// covered by a 16B-aligned 336B window of 21 B128 transfers; reads apply a
// shift of (rowIdx & 3) floats. sumexp(row) is then computed with
// V_WMMA_F32_16X16X32_F16 (B = ones; 3 K-steps cover 81 classes padded to 96).
__global__ void k3_conf(const float* __restrict__ conf,
                        const int*   __restrict__ confT,
                        float* __restrict__ nllArr,
                        float* __restrict__ mineArr,
                        float* lossAcc) {
    // 88-float row slots: 84 window floats + shift slack, 352B stride (16B mult.)
    __shared__ __align__(16) float sc[8][16][88];
    int b = blockIdx.y;
    int lane = threadIdx.x & 31;
    int wave = threadIdx.x >> 5;
    int pbase = blockIdx.x * 128 + wave * 16;   // 8 waves * 16 priors

    // ---- async staging: 16 rows x 21 B128 chunks, round-robin over lanes ----
    for (int idx = lane; idx < 16 * 21; idx += 32) {
        int row = idx / 21, chunk = idx % 21;
        int pr = pbase + row;
        int prc = (pr < PP) ? pr : PP - 1;
        size_t rowByte = ((size_t)b * PP + prc) * (size_t)(NC * 4);
        size_t alignedByte = rowByte & ~(size_t)15;         // 16B-aligned window
        unsigned long long ga =
            (unsigned long long)((const char*)conf + alignedByte + (size_t)chunk * 16);
        unsigned ldsOff =
            (unsigned)(unsigned long long)(const void*)&sc[wave][row][chunk * 4];
        asm volatile("global_load_async_to_lds_b128 %0, %1, off"
                     :: "v"(ldsOff), "v"(ga) : "memory");
    }
    asm volatile("s_wait_asynccnt 0x0" ::: "memory");

    int m = lane & 15;                           // row owned by this lane
    int p = pbase + m;
    int pc = (p < PP) ? p : PP - 1;
    int shift = (int)(((size_t)b * PP + pc) & 3);   // 324 % 16 == 4 -> rowIdx&3 floats
    const float* cr = &sc[wave][m][shift];

    float rmax = -1e30f;
    for (int k = 0; k < NC; ++k) rmax = fmaxf(rmax, cr[k]);

    V16H bm;
    for (int h = 0; h < 16; ++h) bm.e[h] = (_Float16)1.0f;
    V8F acc;
    for (int r = 0; r < 8; ++r) acc.e[r] = 0.0f;

    for (int k0 = 0; k0 < 96; k0 += 32) {
        V16H am;
        for (int h = 0; h < 16; ++h) {
            // 16-bit A 16x32 layout: lanes<16: K=h (h<8), h+8 (h>=8); lanes>=16: +8
            int K = h + ((h >= 8) ? 8 : 0) + ((lane >= 16) ? 8 : 0);
            int k = k0 + K;
            float e = (k < NC) ? expf(cr[k] - rmax) : 0.0f;
            am.e[h] = (_Float16)e;
        }
        acc.v = __builtin_amdgcn_wmma_f32_16x16x32_f16(
            /*neg_a=*/false, am.v, /*neg_b=*/false, bm.v,
            /*c_mod=*/(short)0, acc.v, /*reuse_a=*/false, /*reuse_b=*/false);
    }
    // C/D layout: lane l<16 holds (M=r, N=l); l>=16 holds (M=r+8, N=l-16).
    // All columns equal rowsum -> fetch row m from lane (m&7)+16*[m>=8].
    float local = acc.e[lane & 7];
    int src = (m & 7) + ((m & 8) ? 16 : 0);
    float sumexp = __shfl(local, src, 32);

    if (lane < 16 && p < PP) {
        size_t i = (size_t)b * PP + p;
        int ctv = confT[i];
        int tgt = (ctv > 0) ? ctv : 0;
        float nll = -(cr[tgt] - rmax - logf(sumexp));
        nllArr[i]  = nll;
        mineArr[i] = (ctv != 0) ? 0.0f : nll;      // exclude pos & neutral
        if (ctv > 0) atomicAdd(lossAcc + 1, nll);  // positive part of conf loss
    }
}

// ---------------- K4: per-batch histogram of mine float bits ----------------
__global__ void k4_hist(const float* __restrict__ mineArr, unsigned int* hist) {
    __shared__ unsigned int lh[1024];
    int b = blockIdx.y, t = threadIdx.x;
    for (int i = t; i < 1024; i += 256) lh[i] = 0u;
    __syncthreads();
    int p = blockIdx.x * 256 + t;
    if (p < PP) {
        unsigned bits = __float_as_uint(mineArr[(size_t)b * PP + p]); // >=0 -> monotonic
        atomicAdd(&lh[bits >> 21], 1u);
    }
    __syncthreads();
    for (int i = t; i < 1024; i += 256)
        if (lh[i]) atomicAdd(&hist[b * 1024 + i], lh[i]);
}

// ---------------- K5: find boundary bin + in-bin quota per batch ----------------
__global__ void k5_select(const unsigned int* __restrict__ hist,
                          const int* __restrict__ nPos,
                          int* boundT, int* quota) {
    int b = threadIdx.x;
    if (b >= BB) return;
    long long nn = 3LL * (long long)nPos[b];
    if (nn > PP - 1) nn = PP - 1;
    int T = 1024; long long q = 0;
    if (nn > 0) {
        long long cum = 0;
        for (int bin = 1023; bin >= 0; --bin) {
            unsigned h = hist[b * 1024 + bin];
            if (cum + (long long)h >= nn) { T = bin; q = nn - cum; break; }
            cum += h;
        }
    }
    boundT[b] = T; quota[b] = (int)q;
}

// ---------------- K6: negative selection + conf loss ----------------
__global__ void k6_neg(const float* __restrict__ mineArr,
                       const float* __restrict__ nllArr,
                       const int*   __restrict__ confT,
                       const int* __restrict__ boundT,
                       const int* __restrict__ quota,
                       float* lossAcc) {
    int b = blockIdx.y, t = threadIdx.x;
    int p = blockIdx.x * 256 + t;
    if (p >= PP) return;
    int T = boundT[b], q = quota[b];
    size_t i = (size_t)b * PP + p;
    unsigned bits = __float_as_uint(mineArr[i]);
    int bin = (int)(bits >> 21);
    bool sel = false;
    if (bin > T) {
        sel = true;
    } else if (bin == T && q > 0) {
        // Exact stable rank within boundary bin: (value desc, index asc)
        const float* mb = mineArr + (size_t)b * PP;
        float myv = __uint_as_float(bits);
        int cnt = 0;
        for (int j = 0; j < PP && cnt < q; ++j) {
            unsigned jb = __float_as_uint(mb[j]);
            if ((int)(jb >> 21) == T) {
                float vj = __uint_as_float(jb);
                if (vj > myv || (vj == myv && j < p)) ++cnt;
            }
        }
        sel = (cnt < q);
    }
    if (sel && confT[i] == 0) atomicAdd(lossAcc + 1, nllArr[i]);
}

// ---------------- K7: finalize ----------------
__global__ void k7_fin(const float* lossAcc, const int* Ntot, float* out) {
    if (threadIdx.x == 0 && blockIdx.x == 0) {
        int n = *Ntot; if (n < 1) n = 1;
        float N = (float)n;
        out[0] = lossAcc[0] / N;
        out[1] = lossAcc[1] / N;
    }
}

extern "C" void kernel_launch(void* const* d_in, const int* in_sizes, int n_in,
                              void* d_out, int out_size, void* d_ws, size_t ws_size,
                              hipStream_t stream) {
    const float* loc    = (const float*)d_in[0];   // [B,P,4]
    const float* conf   = (const float*)d_in[1];   // [B,P,81]
    const float* priors = (const float*)d_in[2];   // [P,4]
    const float* gtb    = (const float*)d_in[3];   // [B,32,4]
    const int*   gtl    = (const int*)d_in[4];     // [B,32]
    float* out = (float*)d_out;

    const size_t BP = (size_t)BB * PP;
    char* w = (char*)d_ws;
    float* lossAcc              = (float*)(w + 0);
    int*   nPos                 = (int*)(w + 64);
    int*   Ntot                 = (int*)(w + 128);
    unsigned long long* bestPk  = (unsigned long long*)(w + 192);
    size_t off = 192 + (size_t)BB * NOBJ * 8;
    float* btOv   = (float*)(w + off); off += BP * 4;
    int*   btIdx  = (int*)  (w + off); off += BP * 4;
    int*   confT  = (int*)  (w + off); off += BP * 4;
    float* nllArr = (float*)(w + off); off += BP * 4;
    float* mineAr = (float*)(w + off); off += BP * 4;
    unsigned int* hist = (unsigned int*)(w + off); off += (size_t)BB * 1024 * 4;
    int* boundT = (int*)(w + off); off += 64;
    int* quota  = (int*)(w + off); off += 64;

    dim3 gP((PP + 255) / 256, BB);
    dim3 gW((PP + 127) / 128, BB);

    k0_init<<<dim3((BB * 1024 + 255) / 256), dim3(256), 0, stream>>>(
        lossAcc, nPos, Ntot, bestPk, hist);
    k1_match<<<gP, dim3(256), 0, stream>>>(priors, gtb, btOv, btIdx, bestPk);
    k2_loc<<<gP, dim3(256), 0, stream>>>(loc, priors, gtb, gtl, btOv, btIdx,
                                         bestPk, confT, lossAcc, nPos, Ntot);
    k3_conf<<<gW, dim3(256), 0, stream>>>(conf, confT, nllArr, mineAr, lossAcc);
    k4_hist<<<gP, dim3(256), 0, stream>>>(mineAr, hist);
    k5_select<<<dim3(1), dim3(32), 0, stream>>>(hist, nPos, boundT, quota);
    k6_neg<<<gP, dim3(256), 0, stream>>>(mineAr, nllArr, confT, boundT, quota, lossAcc);
    k7_fin<<<dim3(1), dim3(64), 0, stream>>>(lossAcc, Ntot, out);
}
